// RSSM_50440095924395
// MI455X (gfx1250) — compile-verified, matrix operand order checked by reading, not run on hardware
//
#include <hip/hip_runtime.h>

// ---------------------------------------------------------------------------
// RSSM observe scan for MI455X (gfx1250, wave32, WMMA).
// B=256, T=64, A=32, E=1024, D=1024, S=128.
// Strategy: batch rows are recurrence-independent -> 16 workgroups each own a
// 16-row batch tile and run all 64 steps privately. All matmuls use
// v_wmma_f32_16x16x32_bf16 (f32 accumulate). Weights are packed once into
// WMMA-B fragment layout (bf16); h lives in f32 registers (C layout) and as
// bf16 in LDS for the A operand. embed projections are hoisted into a fully
// parallel WMMA GEMM before the scan. Phase-1 processes d-tiles in pairs so
// each LDS A-fragment load feeds 6 WMMAs (3 gates x 2 tiles).
// ---------------------------------------------------------------------------

typedef __attribute__((ext_vector_type(16))) __bf16 v16bf;
typedef __attribute__((ext_vector_type(8)))  float  v8f;

#define NB   256
#define NT   64
#define NA   32
#define NE   1024
#define ND   1024
#define NS   128
#define G3   (3 * ND)          // 3072
#define KIH  (NS + NA)         // 160

__device__ __forceinline__ v8f wmma_bf16(v16bf a, v16bf b, v8f c) {
    return __builtin_amdgcn_wmma_f32_16x16x32_bf16(
        /*neg_a=*/false, a, /*neg_b=*/false, b,
        /*c_mod=*/(short)0, c, /*reuse_a=*/false, /*reuse_b=*/false);
}

// B-matrix fragment: frag index * 512 halves, lane * 16 halves (32B/lane).
__device__ __forceinline__ v16bf ldB(const __bf16* base, int frag, int lane) {
    return *(const v16bf*)(base + (((size_t)frag) << 9) + (lane << 4));
}

__device__ __forceinline__ float fsigmoid(float x) {
    return 1.0f / (1.0f + __expf(-x));
}

// ---------------------------------------------------------------------------
// K0a: pack a row-major f32 matrix (N x rowStride), logical B[k][n] =
// src[n*rowStride + colBase + k], into bf16 WMMA-B fragment layout:
// frag f = nt*numKt + kt ; within frag: lane(32) x 16 halves,
// k = kt*32 + (lane>>4)*16 + i ; n = nt*16 + (lane&15).
// ---------------------------------------------------------------------------
__global__ __launch_bounds__(256) void pack_b_kernel(
    const float* __restrict__ src, __bf16* __restrict__ dst,
    int numKt, int rowStride, int colBase, int totalHalves)
{
    int tid = blockIdx.x * 256 + threadIdx.x;
    if (tid >= totalHalves) return;
    int i  = tid & 15;
    int l  = (tid >> 4) & 31;
    int f  = tid >> 9;
    int nt = f / numKt;
    int kt = f - nt * numKt;
    int k  = kt * 32 + (l >> 4) * 16 + i;
    int n  = nt * 16 + (l & 15);
    dst[tid] = (__bf16)src[(size_t)n * rowStride + colBase + k];
}

// K0b: pre-sum GRU biases: bsum[0:2048]=b_ih+b_hh (r,u); b_in/b_hn for n gate.
__global__ __launch_bounds__(256) void bias_kernel(
    const float* __restrict__ b_ih, const float* __restrict__ b_hh,
    float* __restrict__ bsum, float* __restrict__ b_in, float* __restrict__ b_hn)
{
    int i = blockIdx.x * 256 + threadIdx.x;
    if (i < 2 * ND) bsum[i] = b_ih[i] + b_hh[i];
    if (i < ND) { b_in[i] = b_ih[2 * ND + i]; b_hn[i] = b_hh[2 * ND + i]; }
}

// ---------------------------------------------------------------------------
// K1: e_m/e_l = embed @ Wq{m,l}[:,D:].T + b  (B*T=16384 rows, K=1024, N=128).
// One block per 16-row M tile; 8 waves, wave w owns s-tile w for both mats.
// ---------------------------------------------------------------------------
__global__ __launch_bounds__(256) void emb_proj_kernel(
    const float* __restrict__ embed,
    const __bf16* __restrict__ WqmeB, const __bf16* __restrict__ WqleB,
    const float* __restrict__ b_qm, const float* __restrict__ b_ql,
    float* __restrict__ e_m, float* __restrict__ e_l)
{
    const int lane = threadIdx.x & 31;
    const int wave = threadIdx.x >> 5;
    const int mt   = blockIdx.x;                 // 0..1023
    const int m    = lane & 15;
    const int kh   = lane >> 4;
    const size_t rowbase = (size_t)(mt * 16 + m) * NE;

    v8f am = {}; v8f al = {};
#pragma unroll 4
    for (int kt = 0; kt < NE / 32; ++kt) {
        const float* src = embed + rowbase + kt * 32 + kh * 16;
        v16bf a;
#pragma unroll
        for (int i = 0; i < 16; ++i) a[i] = (__bf16)src[i];
        v16bf bm = ldB(WqmeB, wave * 32 + kt, lane);
        v16bf bl = ldB(WqleB, wave * 32 + kt, lane);
        am = wmma_bf16(a, bm, am);
        al = wmma_bf16(a, bl, al);
    }
    const int scol  = wave * 16 + (lane & 15);
    const int rbase = (lane >> 4) * 8;
    const float bm_ = b_qm[scol];
    const float bl_ = b_ql[scol];
#pragma unroll
    for (int i = 0; i < 8; ++i) {
        size_t idx = (size_t)(mt * 16 + rbase + i) * NS + scol;
        e_m[idx] = am[i] + bm_;
        e_l[idx] = al[i] + bl_;
    }
}

// ---------------------------------------------------------------------------
// K2: the sequential scan. 16 blocks x 512 threads (16 waves).
// ---------------------------------------------------------------------------
__global__ __launch_bounds__(512) void rssm_scan_kernel(
    const float* __restrict__ actions, const float* __restrict__ deter0,
    const float* __restrict__ stoch0,
    const float* __restrict__ eps_prior, const float* __restrict__ eps_post,
    const __bf16* __restrict__ WihB, const __bf16* __restrict__ WhhB,
    const __bf16* __restrict__ WpmB, const __bf16* __restrict__ WplB,
    const __bf16* __restrict__ WqmB, const __bf16* __restrict__ WqlB,
    const float* __restrict__ bsum, const float* __restrict__ b_in,
    const float* __restrict__ b_hn,
    const float* __restrict__ b_pm, const float* __restrict__ b_pl,
    const float* __restrict__ e_m, const float* __restrict__ e_l,
    float* __restrict__ o_deter, float* __restrict__ o_qm,
    float* __restrict__ o_ql, float* __restrict__ o_zq,
    float* __restrict__ o_pm, float* __restrict__ o_pl,
    float* __restrict__ o_zp)
{
    __shared__ __bf16 hbuf[16 * ND];      // 32 KB: h as bf16 (A operand)
    __shared__ __bf16 za[16 * KIH];       // 5 KB : [z | a] as bf16 (A operand)

    const int lane   = threadIdx.x & 31;
    const int wave   = threadIdx.x >> 5;  // 0..15
    const int batch0 = blockIdx.x * 16;
    const int m      = lane & 15;
    const int kh     = lane >> 4;
    const int rbase  = kh * 8;            // rows m = rbase..rbase+7 in C layout

    // ---- init: h0 (f32 regs + bf16 LDS), z0, a0 ----
    v8f hreg[4];
#pragma unroll
    for (int q = 0; q < 4; ++q) {
        int dt  = wave * 4 + q;
        int col = dt * 16 + m;
        v8f h;
#pragma unroll
        for (int i = 0; i < 8; ++i) {
            float v = deter0[(size_t)(batch0 + rbase + i) * ND + col];
            h[i] = v;
            hbuf[(rbase + i) * ND + col] = (__bf16)v;
        }
        hreg[q] = h;
    }
    if (wave >= 8) {
        int scol = (wave - 8) * 16 + m;
#pragma unroll
        for (int i = 0; i < 8; ++i)
            za[(rbase + i) * KIH + scol] =
                (__bf16)stoch0[(size_t)(batch0 + rbase + i) * NS + scol];
    } else {
#pragma unroll
        for (int rr = 0; rr < 2; ++rr) {
            int row = wave * 2 + rr;
            za[row * KIH + NS + lane] =
                (__bf16)actions[((size_t)(batch0 + row) * NT + 0) * NA + lane];
        }
    }
    __syncthreads();

    for (int t = 0; t < NT; ++t) {
        // ===== phase 1: gates gi+gh, GRU update. d-tiles processed in
        // pairs so one LDS A-fragment feeds 6 WMMAs (3 gates x 2 tiles). =====
#pragma unroll
        for (int p = 0; p < 2; ++p) {
            const int d0 = wave * 4 + 2 * p;
            v8f accR[2]  = {};
            v8f accU[2]  = {};
            v8f accIN[2] = {};
            v8f accHN[2] = {};

            // gi: A = [z|a] (K=160)
#pragma unroll
            for (int kt = 0; kt < KIH / 32; ++kt) {
                v16bf a = *(const v16bf*)&za[m * KIH + kt * 32 + kh * 16];
#pragma unroll
                for (int j = 0; j < 2; ++j) {
                    const int dt = d0 + j;
                    v16bf br = ldB(WihB, (0 * 64 + dt) * (KIH / 32) + kt, lane);
                    v16bf bu = ldB(WihB, (1 * 64 + dt) * (KIH / 32) + kt, lane);
                    v16bf bn = ldB(WihB, (2 * 64 + dt) * (KIH / 32) + kt, lane);
                    accR[j]  = wmma_bf16(a, br, accR[j]);
                    accU[j]  = wmma_bf16(a, bu, accU[j]);
                    accIN[j] = wmma_bf16(a, bn, accIN[j]);
                }
            }
            // gh: A = h (K=1024)
#pragma unroll 2
            for (int kt = 0; kt < ND / 32; ++kt) {
                v16bf a = *(const v16bf*)&hbuf[m * ND + kt * 32 + kh * 16];
                if ((kt & 7) == 0) {
#pragma unroll
                    for (int g = 0; g < 3; ++g)
                        __builtin_prefetch(
                            WhhB +
                                (((size_t)((g * 64 + d0) * 32 + kt + 8)) << 9) +
                                (lane << 4), 0, 1);
                }
#pragma unroll
                for (int j = 0; j < 2; ++j) {
                    const int dt = d0 + j;
                    v16bf br = ldB(WhhB, (0 * 64 + dt) * 32 + kt, lane);
                    v16bf bu = ldB(WhhB, (1 * 64 + dt) * 32 + kt, lane);
                    v16bf bn = ldB(WhhB, (2 * 64 + dt) * 32 + kt, lane);
                    accR[j]  = wmma_bf16(a, br, accR[j]);
                    accU[j]  = wmma_bf16(a, bu, accU[j]);
                    accHN[j] = wmma_bf16(a, bn, accHN[j]);
                }
            }
            // GRU elementwise (per-lane column is constant across the 8 rows)
#pragma unroll
            for (int j = 0; j < 2; ++j) {
                const int dt  = d0 + j;
                const int q   = 2 * p + j;
                const int col = dt * 16 + m;
                const float br_  = bsum[col];
                const float bu_  = bsum[ND + col];
                const float bin_ = b_in[col];
                const float bhn_ = b_hn[col];
                v8f h = hreg[q];
                v8f hnew;
#pragma unroll
                for (int i = 0; i < 8; ++i) {
                    float r = fsigmoid(accR[j][i] + br_);
                    float u = fsigmoid(accU[j][i] + bu_);
                    float n = tanhf((accIN[j][i] + bin_) +
                                    r * (accHN[j][i] + bhn_));
                    hnew[i] = (1.0f - u) * n + u * h[i];
                }
                hreg[q] = hnew;
#pragma unroll
                for (int i = 0; i < 8; ++i) {
                    size_t idx =
                        ((size_t)(batch0 + rbase + i) * NT + t) * ND + col;
                    o_deter[idx] = hnew[i];
                }
            }
        }
        __syncthreads();   // all reads of hbuf (h_old) done

        // publish h_new (bf16) for phase 2 and next step
#pragma unroll
        for (int q = 0; q < 4; ++q) {
            int col = (wave * 4 + q) * 16 + m;
#pragma unroll
            for (int i = 0; i < 8; ++i)
                hbuf[(rbase + i) * ND + col] = (__bf16)hreg[q][i];
        }
        __syncthreads();   // hbuf now holds h_new

        // ===== phase 2: heads. waves 0-7 prior (pm,pl), 8-15 posterior =====
        {
            const bool prior = (wave < 8);
            const int st = prior ? wave : wave - 8;
            const __bf16* Bm = prior ? WpmB : WqmB;
            const __bf16* Bl = prior ? WplB : WqlB;
            v8f am = {}, al = {};
#pragma unroll 4
            for (int kt = 0; kt < ND / 32; ++kt) {
                v16bf a = *(const v16bf*)&hbuf[m * ND + kt * 32 + kh * 16];
                v16bf bm = ldB(Bm, st * 32 + kt, lane);
                v16bf bl = ldB(Bl, st * 32 + kt, lane);
                am = wmma_bf16(a, bm, am);
                al = wmma_bf16(a, bl, al);
            }
            const int scol = st * 16 + m;
            if (prior) {
                const float bm_ = b_pm[scol];
                const float bl_ = b_pl[scol];
#pragma unroll
                for (int i = 0; i < 8; ++i) {
                    size_t idx =
                        ((size_t)(batch0 + rbase + i) * NT + t) * NS + scol;
                    float pm = am[i] + bm_;
                    float pl = al[i] + bl_;
                    float zp = pm + eps_prior[idx] * __expf(pl);
                    o_pm[idx] = pm; o_pl[idx] = pl; o_zp[idx] = zp;
                }
                // stage a_{t+1} into za (waves 0-7, 2 rows each, lane = col)
                if (t + 1 < NT) {
#pragma unroll
                    for (int rr = 0; rr < 2; ++rr) {
                        int row = wave * 2 + rr;
                        float v = actions[((size_t)(batch0 + row) * NT +
                                           (t + 1)) * NA + lane];
                        za[row * KIH + NS + lane] = (__bf16)v;
                    }
                }
            } else {
#pragma unroll
                for (int i = 0; i < 8; ++i) {
                    size_t idx =
                        ((size_t)(batch0 + rbase + i) * NT + t) * NS + scol;
                    float qm = am[i] + e_m[idx];   // e_m includes b_qm
                    float ql = al[i] + e_l[idx];   // e_l includes b_ql
                    float zq = qm + eps_post[idx] * __expf(ql);
                    o_qm[idx] = qm; o_ql[idx] = ql; o_zq[idx] = zq;
                    za[(rbase + i) * KIH + scol] = (__bf16)zq;  // carry z
                }
            }
        }
        __syncthreads();   // za ready for next step
    }
}

// ---------------------------------------------------------------------------
// host launcher
// ---------------------------------------------------------------------------
extern "C" void kernel_launch(void* const* d_in, const int* in_sizes, int n_in,
                              void* d_out, int out_size, void* d_ws, size_t ws_size,
                              hipStream_t stream) {
    const float* embed   = (const float*)d_in[0];
    const float* actions = (const float*)d_in[1];
    const float* deter0  = (const float*)d_in[2];
    const float* stoch0  = (const float*)d_in[3];
    const float* eps_p   = (const float*)d_in[4];
    const float* eps_q   = (const float*)d_in[5];
    const float* W_ih    = (const float*)d_in[6];
    const float* W_hh    = (const float*)d_in[7];
    const float* b_ih    = (const float*)d_in[8];
    const float* b_hh    = (const float*)d_in[9];
    const float* W_pm    = (const float*)d_in[10];
    const float* b_pm    = (const float*)d_in[11];
    const float* W_pl    = (const float*)d_in[12];
    const float* b_pl    = (const float*)d_in[13];
    const float* W_qm    = (const float*)d_in[14];
    const float* b_qm    = (const float*)d_in[15];
    const float* W_ql    = (const float*)d_in[16];
    const float* b_ql    = (const float*)d_in[17];

    // ---- workspace carve-up (all offsets 256B aligned), ~25.6 MB total ----
    char* ws = (char*)d_ws;
    __bf16* WhhB  = (__bf16*)ws; ws += (size_t)G3 * ND * 2;       // 6 MB
    __bf16* WihB  = (__bf16*)ws; ws += (size_t)G3 * KIH * 2;      // 0.94 MB
    __bf16* WpmB  = (__bf16*)ws; ws += (size_t)NS * ND * 2;
    __bf16* WplB  = (__bf16*)ws; ws += (size_t)NS * ND * 2;
    __bf16* WqmB  = (__bf16*)ws; ws += (size_t)NS * ND * 2;
    __bf16* WqlB  = (__bf16*)ws; ws += (size_t)NS * ND * 2;
    __bf16* WqmeB = (__bf16*)ws; ws += (size_t)NS * NE * 2;
    __bf16* WqleB = (__bf16*)ws; ws += (size_t)NS * NE * 2;
    float*  bsum  = (float*)ws;  ws += 2 * ND * 4;
    float*  b_in  = (float*)ws;  ws += ND * 4;
    float*  b_hn  = (float*)ws;  ws += ND * 4;
    float*  e_m   = (float*)ws;  ws += (size_t)NB * NT * NS * 4;  // 8 MB
    float*  e_l   = (float*)ws;  ws += (size_t)NB * NT * NS * 4;  // 8 MB

    // ---- output carve-up (tuple order) ----
    float* out     = (float*)d_out;
    float* o_deter = out;                                 // (B,T,D)
    float* o_qm    = o_deter + (size_t)NB * NT * ND;      // post_mean
    float* o_ql    = o_qm    + (size_t)NB * NT * NS;      // post_logstd
    float* o_zq    = o_ql    + (size_t)NB * NT * NS;      // post_stoch
    float* o_pm    = o_zq    + (size_t)NB * NT * NS;      // prior_mean
    float* o_pl    = o_pm    + (size_t)NB * NT * NS;      // prior_logstd
    float* o_zp    = o_pl    + (size_t)NB * NT * NS;      // prior_stoch

    // ---- K0: pack weights to bf16 WMMA-B fragments ----
    auto packGrid = [](int halves) { return dim3((halves + 255) / 256); };
    pack_b_kernel<<<packGrid(G3 * ND), 256, 0, stream>>>(W_hh, WhhB, ND / 32, ND, 0, G3 * ND);
    pack_b_kernel<<<packGrid(G3 * KIH), 256, 0, stream>>>(W_ih, WihB, KIH / 32, KIH, 0, G3 * KIH);
    pack_b_kernel<<<packGrid(NS * ND), 256, 0, stream>>>(W_pm, WpmB, ND / 32, ND, 0, NS * ND);
    pack_b_kernel<<<packGrid(NS * ND), 256, 0, stream>>>(W_pl, WplB, ND / 32, ND, 0, NS * ND);
    pack_b_kernel<<<packGrid(NS * ND), 256, 0, stream>>>(W_qm, WqmB, ND / 32, ND + NE, 0, NS * ND);
    pack_b_kernel<<<packGrid(NS * ND), 256, 0, stream>>>(W_ql, WqlB, ND / 32, ND + NE, 0, NS * ND);
    pack_b_kernel<<<packGrid(NS * NE), 256, 0, stream>>>(W_qm, WqmeB, NE / 32, ND + NE, ND, NS * NE);
    pack_b_kernel<<<packGrid(NS * NE), 256, 0, stream>>>(W_ql, WqleB, NE / 32, ND + NE, ND, NS * NE);
    bias_kernel<<<dim3(8), 256, 0, stream>>>(b_ih, b_hh, bsum, b_in, b_hn);

    // ---- K1: hoisted embed projections (fully parallel WMMA GEMM) ----
    emb_proj_kernel<<<dim3(NB * NT / 16), 256, 0, stream>>>(
        embed, WqmeB, WqleB, b_qm, b_ql, e_m, e_l);

    // ---- K2: the scan (16 batch tiles x 16 waves) ----
    rssm_scan_kernel<<<dim3(NB / 16), 512, 0, stream>>>(
        actions, deter0, stoch0, eps_p, eps_q,
        WihB, WhhB, WpmB, WplB, WqmB, WqlB,
        bsum, b_in, b_hn, b_pm, b_pl, e_m, e_l,
        o_deter, o_qm, o_ql, o_zq, o_pm, o_pl, o_zp);
}
